// MultiHeadedAttention_72988674228706
// MI455X (gfx1250) — compile-verified, matrix operand order checked by reading, not run on hardware
//
#include <hip/hip_runtime.h>

#define D_MODEL   768
#define NUM_HEADS 12
#define HEAD      64
#define BATCH     4
#define SEQ       2048
#define M_TOTAL   (BATCH * SEQ)   // 8192

typedef _Float16 half16 __attribute__((ext_vector_type(16)));
typedef _Float16 half8  __attribute__((ext_vector_type(8)));
typedef _Float16 half4  __attribute__((ext_vector_type(4)));
typedef float    float8 __attribute__((ext_vector_type(8)));

static __device__ __forceinline__ float8 wmma16(half16 a, half16 b, float8 c) {
  // D = A(16x32 f16) * B(32x16 f16) + C(16x16 f32)
  return __builtin_amdgcn_wmma_f32_16x16x32_f16(false, a, false, b, (short)0, c,
                                                false, false);
}

// Load a 16x32 f16 tile (rows x K) in WMMA A/B register layout from a
// row-major matrix with leading dimension `ld` (halves).
static __device__ __forceinline__ half16 load_tile_h(const _Float16* base, int ld) {
  int lane = threadIdx.x & 31;
  int row  = lane & 15;
  int cb   = (lane >> 4) << 3;   // 0 or 8
  const _Float16* p = base + (size_t)row * ld + cb;
  union { half16 v; half8 h[2]; } u;
  u.h[0] = *(const half8*)(p);        // K cb..cb+7
  u.h[1] = *(const half8*)(p + 16);   // K cb+16..cb+23
  return u.v;
}

// Same layout from f32 row-major, converting to f16.
static __device__ __forceinline__ half16 load_tile_f(const float* base, int ld) {
  int lane = threadIdx.x & 31;
  int row  = lane & 15;
  int cb   = (lane >> 4) << 3;
  const float* p = base + (size_t)row * ld + cb;
  union { half16 v; _Float16 e[16]; } u;
#pragma unroll
  for (int i = 0; i < 8; ++i) {
    u.e[i]     = (_Float16)p[i];
    u.e[i + 8] = (_Float16)p[i + 16];
  }
  return u.v;
}

// ---------------------------------------------------------------------------
// Kernel 0: bulk f32 -> f16 convert (for weight matrices).
// ---------------------------------------------------------------------------
__global__ __launch_bounds__(256)
void cvt_f16_kernel(const float* __restrict__ src, _Float16* __restrict__ dst,
                    int n4) {
  int i = blockIdx.x * 256 + threadIdx.x;
  if (i < n4) {
    float4 v = ((const float4*)src)[i];
    half4 h = { (_Float16)v.x, (_Float16)v.y, (_Float16)v.z, (_Float16)v.w };
    ((half4*)dst)[i] = h;
  }
}

// ---------------------------------------------------------------------------
// Kernel 1: Y = X @ W^T + bias (f16 W, f32 X converted in-loop).
// 4x4 register blocking: one wave computes a 64x64 output block.
// transpose_out == 0 : [B][H][S][d]  (Q, K);  == 1 : [B][H][d][S]  (V^T)
// Grid: (M_TOTAL/64, D_MODEL/64) = (128, 12), block 32.
// ---------------------------------------------------------------------------
__global__ __launch_bounds__(32)
void proj_qkv_kernel(const float* __restrict__ X, const _Float16* __restrict__ Wh,
                     const float* __restrict__ bias, _Float16* __restrict__ Y,
                     int transpose_out) {
  const int m0 = blockIdx.x * 64;
  const int n0 = blockIdx.y * 64;
  float8 acc[4][4];
#pragma unroll
  for (int i = 0; i < 4; ++i)
#pragma unroll
    for (int j = 0; j < 4; ++j) acc[i][j] = (float8){};

  for (int kk = 0; kk < D_MODEL; kk += 32) {
    half16 a[4], bt[4];
#pragma unroll
    for (int i = 0; i < 4; ++i)
      a[i]  = load_tile_f(X  + (size_t)(m0 + 16*i) * D_MODEL + kk, D_MODEL);
#pragma unroll
    for (int j = 0; j < 4; ++j)
      bt[j] = load_tile_h(Wh + (size_t)(n0 + 16*j) * D_MODEL + kk, D_MODEL);
#pragma unroll
    for (int i = 0; i < 4; ++i)
#pragma unroll
      for (int j = 0; j < 4; ++j)
        acc[i][j] = wmma16(a[i], bt[j], acc[i][j]);
  }

  const int lane = threadIdx.x & 31;
  const int N  = lane & 15;
  const int Mb = (lane >> 4) << 3;
#pragma unroll
  for (int j = 0; j < 4; ++j) {
    int n  = n0 + 16*j + N;
    float bv = bias[n];
    int h  = n >> 6;
    int dc = n & 63;
#pragma unroll
    for (int i = 0; i < 4; ++i) {
#pragma unroll
      for (int r = 0; r < 8; ++r) {
        int mm = m0 + 16*i + Mb + r;
        int b_ = mm >> 11;            // / SEQ
        int s  = mm & (SEQ - 1);
        float v = acc[i][j][r] + bv;
        size_t idx = transpose_out
          ? (((size_t)b_ * NUM_HEADS + h) * HEAD + dc) * SEQ + s
          : (((size_t)b_ * NUM_HEADS + h) * SEQ + s) * HEAD + dc;
        Y[idx] = (_Float16)v;
      }
    }
  }
}

// ---------------------------------------------------------------------------
// Kernel 2: causal flash attention, one wave per (b, h, 32-query tile).
// Two 16-row M-tiles share every K/V tile load.
// Grid: (SEQ/32, BATCH*NUM_HEADS) = (64, 48), block 32.
// ---------------------------------------------------------------------------
__global__ __launch_bounds__(32)
void attn_kernel(const _Float16* __restrict__ Qh, const _Float16* __restrict__ Kh,
                 const _Float16* __restrict__ Vt, _Float16* __restrict__ Ctx) {
  __shared__ __align__(16) _Float16 Pl[2][16 * 32];

  int q0 = blockIdx.x * 32;
  int bh = blockIdx.y;
  int b  = bh / NUM_HEADS;
  int h  = bh % NUM_HEADS;

  const _Float16* qbase = Qh + (size_t)bh * SEQ * HEAD;
  const _Float16* kbase = Kh + (size_t)bh * SEQ * HEAD;
  const _Float16* vbase = Vt + (size_t)bh * HEAD * SEQ;   // [d][S]

  half16 qa[2][2];
#pragma unroll
  for (int mt = 0; mt < 2; ++mt) {
    qa[mt][0] = load_tile_h(qbase + (size_t)(q0 + 16*mt) * HEAD,      HEAD);
    qa[mt][1] = load_tile_h(qbase + (size_t)(q0 + 16*mt) * HEAD + 32, HEAD);
  }

  int lane = threadIdx.x & 31;
  int N  = lane & 15;
  int Mb = (lane >> 4) << 3;

  float m[2][8], l[2][8];
  float8 o[2][4];
#pragma unroll
  for (int mt = 0; mt < 2; ++mt) {
#pragma unroll
    for (int r = 0; r < 8; ++r) { m[mt][r] = -3.0e38f; l[mt][r] = 0.0f; }
#pragma unroll
    for (int t = 0; t < 4; ++t) o[mt][t] = (float8){};
  }

  const int kend = q0 + 32;
  for (int kb = 0; kb < kend; kb += 32) {
    // gfx1250 prefetch of the next K/V blocks (speculative; OOB is dropped)
    __builtin_prefetch(kbase + (size_t)(kb + 32) * HEAD, 0, 1);
    __builtin_prefetch(vbase + kb + 32, 0, 1);

    // ---- K tiles for 32 keys, both d halves (shared by both M-tiles) ------
    half16 kt[2][2];
#pragma unroll
    for (int nt = 0; nt < 2; ++nt) {
      kt[nt][0] = load_tile_h(kbase + (size_t)(kb + 16*nt) * HEAD,      HEAD);
      kt[nt][1] = load_tile_h(kbase + (size_t)(kb + 16*nt) * HEAD + 32, HEAD);
    }
    // ---- scores: 32 queries x 32 keys --------------------------------------
    float8 s[2][2];
#pragma unroll
    for (int mt = 0; mt < 2; ++mt)
#pragma unroll
      for (int nt = 0; nt < 2; ++nt) {
        float8 t0 = wmma16(qa[mt][0], kt[nt][0], (float8){});
        s[mt][nt] = wmma16(qa[mt][1], kt[nt][1], t0);
      }
    // ---- scale + causal mask + online softmax ------------------------------
#pragma unroll
    for (int mt = 0; mt < 2; ++mt) {
#pragma unroll
      for (int r = 0; r < 8; ++r) {
        int q = q0 + 16*mt + Mb + r;
        float v0 = (kb + N      <= q) ? s[mt][0][r] * 0.125f : -3.0e38f;
        float v1 = (kb + 16 + N <= q) ? s[mt][1][r] * 0.125f : -3.0e38f;
        float rm = fmaxf(v0, v1);
        rm = fmaxf(rm, __shfl_xor(rm, 1));
        rm = fmaxf(rm, __shfl_xor(rm, 2));
        rm = fmaxf(rm, __shfl_xor(rm, 4));
        rm = fmaxf(rm, __shfl_xor(rm, 8));
        float mn = fmaxf(m[mt][r], rm);
        float sc = __expf(m[mt][r] - mn);
        float p0 = __expf(v0 - mn);
        float p1 = __expf(v1 - mn);
        float rs = p0 + p1;
        rs += __shfl_xor(rs, 1);
        rs += __shfl_xor(rs, 2);
        rs += __shfl_xor(rs, 4);
        rs += __shfl_xor(rs, 8);
        l[mt][r] = l[mt][r] * sc + rs;
        m[mt][r] = mn;
        o[mt][0][r] *= sc; o[mt][1][r] *= sc;
        o[mt][2][r] *= sc; o[mt][3][r] *= sc;
        Pl[mt][(Mb + r) * 32 + N]      = (_Float16)p0;
        Pl[mt][(Mb + r) * 32 + 16 + N] = (_Float16)p1;
      }
    }
    __syncthreads();                    // single-wave WG -> S_NOP
    // ---- O += P V (V tiles shared by both M-tiles) -------------------------
    half16 pa0 = load_tile_h(Pl[0], 32);
    half16 pa1 = load_tile_h(Pl[1], 32);
#pragma unroll
    for (int t = 0; t < 4; ++t) {
      half16 vt = load_tile_h(vbase + (size_t)(16*t) * SEQ + kb, SEQ);
      o[0][t] = wmma16(pa0, vt, o[0][t]);
      o[1][t] = wmma16(pa1, vt, o[1][t]);
    }
    __syncthreads();
  }

  // ---- normalize + write merged-head context [B][S][D] f16 -----------------
#pragma unroll
  for (int mt = 0; mt < 2; ++mt) {
    _Float16* cbase = Ctx + ((size_t)b * SEQ + q0 + 16*mt) * D_MODEL + h * HEAD;
#pragma unroll
    for (int r = 0; r < 8; ++r) {
      float inv = 1.0f / l[mt][r];
      _Float16* cp = cbase + (size_t)(Mb + r) * D_MODEL + N;
      cp[0]  = (_Float16)(o[mt][0][r] * inv);
      cp[16] = (_Float16)(o[mt][1][r] * inv);
      cp[32] = (_Float16)(o[mt][2][r] * inv);
      cp[48] = (_Float16)(o[mt][3][r] * inv);
    }
  }
}

// ---------------------------------------------------------------------------
// Kernel 3: out = Ctx @ W_O^T + b_O   (all-f16 operands, f32 out).
// 4x4 register blocking. Grid: (128, 12), block 32.
// ---------------------------------------------------------------------------
__global__ __launch_bounds__(32)
void out_proj_kernel(const _Float16* __restrict__ Ctx, const _Float16* __restrict__ Wh,
                     const float* __restrict__ bias, float* __restrict__ out) {
  const int m0 = blockIdx.x * 64;
  const int n0 = blockIdx.y * 64;
  float8 acc[4][4];
#pragma unroll
  for (int i = 0; i < 4; ++i)
#pragma unroll
    for (int j = 0; j < 4; ++j) acc[i][j] = (float8){};

  for (int kk = 0; kk < D_MODEL; kk += 32) {
    half16 a[4], bt[4];
#pragma unroll
    for (int i = 0; i < 4; ++i)
      a[i]  = load_tile_h(Ctx + (size_t)(m0 + 16*i) * D_MODEL + kk, D_MODEL);
#pragma unroll
    for (int j = 0; j < 4; ++j)
      bt[j] = load_tile_h(Wh  + (size_t)(n0 + 16*j) * D_MODEL + kk, D_MODEL);
#pragma unroll
    for (int i = 0; i < 4; ++i)
#pragma unroll
      for (int j = 0; j < 4; ++j)
        acc[i][j] = wmma16(a[i], bt[j], acc[i][j]);
  }

  const int lane = threadIdx.x & 31;
  const int N  = lane & 15;
  const int Mb = (lane >> 4) << 3;
#pragma unroll
  for (int j = 0; j < 4; ++j) {
    float bv = bias[n0 + 16*j + N];
#pragma unroll
    for (int i = 0; i < 4; ++i)
#pragma unroll
      for (int r = 0; r < 8; ++r)
        out[(size_t)(m0 + 16*i + Mb + r) * D_MODEL + n0 + 16*j + N] =
            acc[i][j][r] + bv;
  }
}

// ---------------------------------------------------------------------------
extern "C" void kernel_launch(void* const* d_in, const int* in_sizes, int n_in,
                              void* d_out, int out_size, void* d_ws, size_t ws_size,
                              hipStream_t stream) {
  (void)in_sizes; (void)n_in; (void)out_size; (void)ws_size;
  const float* Xq = (const float*)d_in[0];
  const float* Xk = (const float*)d_in[1];
  const float* Xv = (const float*)d_in[2];
  const float* WQ = (const float*)d_in[3];
  const float* bQ = (const float*)d_in[4];
  const float* WK = (const float*)d_in[5];
  const float* bK = (const float*)d_in[6];
  const float* WV = (const float*)d_in[7];
  const float* bV = (const float*)d_in[8];
  const float* WO = (const float*)d_in[9];
  const float* bO = (const float*)d_in[10];

  const size_t per  = (size_t)BATCH * NUM_HEADS * SEQ * HEAD;  // 6,291,456 halves
  const size_t wsz  = (size_t)D_MODEL * D_MODEL;               //   589,824 halves
  _Float16* Qh  = (_Float16*)d_ws;
  _Float16* Kh  = Qh + per;
  _Float16* Vt  = Kh + per;
  _Float16* Ctx = Vt + per;
  _Float16* WhQ = Ctx + per;
  _Float16* WhK = WhQ + wsz;
  _Float16* WhV = WhK + wsz;
  _Float16* WhO = WhV + wsz;          // total ws use ~55 MB

  dim3 blk(32);
  // weight f32 -> f16 (once per launch; tiny)
  const int n4 = (D_MODEL * D_MODEL) / 4;  // 147,456
  dim3 gCvt((n4 + 255) / 256);
  cvt_f16_kernel<<<gCvt, dim3(256), 0, stream>>>(WQ, WhQ, n4);
  cvt_f16_kernel<<<gCvt, dim3(256), 0, stream>>>(WK, WhK, n4);
  cvt_f16_kernel<<<gCvt, dim3(256), 0, stream>>>(WV, WhV, n4);
  cvt_f16_kernel<<<gCvt, dim3(256), 0, stream>>>(WO, WhO, n4);

  dim3 gProj(M_TOTAL / 64, D_MODEL / 64);   // 128 x 12
  proj_qkv_kernel<<<gProj, blk, 0, stream>>>(Xq, WhQ, bQ, Qh, 0);
  proj_qkv_kernel<<<gProj, blk, 0, stream>>>(Xk, WhK, bK, Kh, 0);
  proj_qkv_kernel<<<gProj, blk, 0, stream>>>(Xv, WhV, bV, Vt, 1);

  dim3 gAttn(SEQ / 32, BATCH * NUM_HEADS);  // 64 x 48
  attn_kernel<<<gAttn, blk, 0, stream>>>(Qh, Kh, Vt, Ctx);

  out_proj_kernel<<<gProj, blk, 0, stream>>>(Ctx, WhO, bO, (float*)d_out);
}
